// MultiHeadSparseAttention_55903294324919
// MI455X (gfx1250) — matrix operand order repaired and checked by reading, not compiled, not run on hardware
//
#include <hip/hip_runtime.h>

// ---------------------------------------------------------------------------
// MultiHeadSparseAttention for MI455X (gfx1250, wave32, WMMA)
//   B=2, H=16, S=2048, D=2048, hd=128, K_KEEP=819
// Pipeline:
//   1) fp32 -> f16 conversion of x, Wq/Wk/Wv, Wo (WMMA operands)
//   2) QKV projection GEMMs via v_wmma_f32_16x16x32_f16 (V stored transposed)
//   3) attention: per 16-query-row block, scores live in LDS (128KB stripe),
//      exact top-k via 4-pass radix select on LDS histograms, softmax,
//      P (f16 in LDS) x V via WMMA, folded 1/sum in epilogue,
//      output stored transposed [B, D, S] as f16
//   4) output projection GEMM (2048^3 per batch) via WMMA + bias, fp32 out
// ---------------------------------------------------------------------------

typedef _Float16 h16 __attribute__((ext_vector_type(16)));
typedef float    f8  __attribute__((ext_vector_type(8)));

#define B_    2
#define H_    16
#define S_    2048
#define D_    2048
#define HD_   128
#define KKEEP 819u
#define NEGV  (-1.0e9f)
#define SCALE 0.08838834764831845f   // 1/sqrt(128)
#define SLD   2052                   // padded fp32 score row (banks: 4*16=64 distinct)
#define PLD   2056                   // padded f16 prob row   (banks: 4*16=64 distinct)

// ---- WMMA fragment loaders (contiguous k = 16h + j, same order for A and B) -------
// A 16x32 f16: lanes 0-15 row=lane, K in {0..15}; lanes 16-31 row=lane-16,
// K in {16..31}.  p = tile origin (row-major, leading dim ld).
__device__ inline h16 ldA(const _Float16* __restrict__ p, long ld, int lane, long kbase) {
  int row = lane & 15, hf = lane >> 4;
  const _Float16* r0 = p + (long)row * ld + kbase + hf * 16;
  h16 a;
#pragma unroll
  for (int j = 0; j < 16; ++j) { a[j] = r0[j]; }
  return a;
}

// B 32x16 f16 with B[k][n] = p[n][k] (p row-major, row n = output column n):
// lane holds column lane&15, 16 contiguous K starting at kbase + (lane>>4)*16.
__device__ inline h16 ldB(const _Float16* __restrict__ p, long ld, int lane, long kbase) {
  int col = lane & 15, hf = lane >> 4;
  const _Float16* r0 = p + (long)col * ld + kbase + hf * 16;
  h16 b;
#pragma unroll
  for (int j = 0; j < 16; ++j) b[j] = r0[j];
  return b;
}

__device__ inline f8 wmma(h16 a, h16 b, f8 c) {
  return __builtin_amdgcn_wmma_f32_16x16x32_f16(false, a, false, b, (short)0, c, false, false);
}

// order-preserving float -> uint key (descending float order == descending key)
__device__ inline unsigned fkey(float f) {
  unsigned u = __float_as_uint(f);
  return (u & 0x80000000u) ? ~u : (u | 0x80000000u);
}

// ---------------------------------------------------------------------------
__global__ void __launch_bounds__(256) cvt_f32_f16(const float* __restrict__ src,
                                                   _Float16* __restrict__ dst, int n) {
  for (int i = blockIdx.x * blockDim.x + threadIdx.x; i < n; i += gridDim.x * blockDim.x)
    dst[i] = (_Float16)src[i];
}

// ---------------------------------------------------------------------------
// QKV projection: one wave per 16-row tile per (b,h). q,k row-major f16; v transposed.
__global__ void __launch_bounds__(128) qkv_proj(const _Float16* __restrict__ x16,
                                                const _Float16* __restrict__ w16,
                                                const float* __restrict__ bq,
                                                const float* __restrict__ bk,
                                                const float* __restrict__ bv,
                                                _Float16* __restrict__ q16,
                                                _Float16* __restrict__ k16,
                                                _Float16* __restrict__ vT16) {
  int lane = threadIdx.x & 31;
  int wid  = blockIdx.x * (blockDim.x >> 5) + (threadIdx.x >> 5);
  int bh = wid / (S_ / 16);
  int mt = wid % (S_ / 16);
  int b = bh / H_, h = bh % H_;
  long qbase = (long)mt * 16;

  const _Float16* Ap = x16 + ((long)b * S_ + qbase) * D_ + (long)h * HD_;
  h16 a[4];
#pragma unroll
  for (int kk = 0; kk < 4; ++kk) a[kk] = ldA(Ap, D_, lane, kk * 32);

  const float* biases[3] = { bq + h * HD_, bk + h * HD_, bv + h * HD_ };
#pragma unroll
  for (int sel = 0; sel < 3; ++sel) {
    const _Float16* W = w16 + ((long)sel * H_ + h) * HD_ * HD_;
    const float* bias = biases[sel];
    for (int nt = 0; nt < 8; ++nt) {
      f8 c = {};
#pragma unroll
      for (int kk = 0; kk < 4; ++kk) {
        h16 bf = ldB(W + (long)nt * 16 * HD_, HD_, lane, kk * 32);
        c = wmma(a[kk], bf, c);
      }
      int n = lane & 15, m0 = (lane >> 4) * 8;
      int e = nt * 16 + n;
      float bb = bias[e];
#pragma unroll
      for (int r = 0; r < 8; ++r) {
        float v = c[r] + bb;
        long srow = qbase + m0 + r;
        if (sel == 0)      q16[((long)bh * S_ + srow) * HD_ + e] = (_Float16)v;
        else if (sel == 1) k16[((long)bh * S_ + srow) * HD_ + e] = (_Float16)v;
        else               vT16[((long)bh * HD_ + e) * S_ + srow] = (_Float16)v;
      }
    }
  }
}

// ---------------------------------------------------------------------------
// Attention: one 256-thread block per (b,h, 16-query-row block).
// Dynamic LDS: sc[16][SLD] fp32 scores, hist[16][256] u32, small row stats,
// p16[16][PLD] f16 probabilities.
__global__ void __launch_bounds__(256) sparse_attn(const _Float16* __restrict__ q16,
                                                   const _Float16* __restrict__ k16,
                                                   const _Float16* __restrict__ vT16,
                                                   _Float16* __restrict__ out_t16) {
  extern __shared__ char smem[];
  float*    sc      = (float*)smem;                                   // 16*SLD fp32
  unsigned* hist    = (unsigned*)(smem + 16 * SLD * 4);               // 16*256 u32
  unsigned* rowPref = (unsigned*)(smem + 16 * SLD * 4 + 16 * 256 * 4);
  unsigned* rowKrem = rowPref + 16;
  float*    rowMax  = (float*)(rowKrem + 16);
  float*    rowSum  = rowMax + 16;
  _Float16* p16     = (_Float16*)(rowSum + 16);                       // 16*PLD f16

  int tid = threadIdx.x;
  int lane = tid & 31, wv = tid >> 5;
  int bh = blockIdx.x / (S_ / 16);
  int mt = blockIdx.x % (S_ / 16);
  int b = bh / H_, h = bh % H_;
  long qbase = (long)mt * 16;

  // ---- phase 1: scores tile [16, 2048] -> LDS (wave wv owns 256 columns) ----
  const _Float16* Qp = q16 + ((long)bh * S_ + qbase) * HD_;
  h16 a[4];
#pragma unroll
  for (int kk = 0; kk < 4; ++kk) a[kk] = ldA(Qp, HD_, lane, kk * 32);

  for (int ct = 0; ct < 16; ++ct) {
    int col0 = wv * 256 + ct * 16;
    f8 c = {};
    const _Float16* Kp = k16 + ((long)bh * S_ + col0) * HD_;
#pragma unroll
    for (int kk = 0; kk < 4; ++kk) {
      h16 bf = ldB(Kp, HD_, lane, kk * 32);
      c = wmma(a[kk], bf, c);
    }
    int n = lane & 15, m0 = (lane >> 4) * 8;
#pragma unroll
    for (int r = 0; r < 8; ++r) {
      int m = m0 + r;
      long grow = qbase + m;
      int col = col0 + n;
      float s = ((long)col <= grow) ? c[r] * SCALE : NEGV;
      sc[m * SLD + col] = s;
    }
  }
  __syncthreads();

  // ---- phase 2: exact top-KKEEP threshold per row (radix select), softmax ---
  int row = tid & 15, sub = tid >> 4;           // 16 threads cooperate per row
  if (sub == 0) { rowPref[row] = 0u; rowKrem[row] = KKEEP; }
  __syncthreads();

  for (int pass = 3; pass >= 0; --pass) {
    for (int i = sub; i < 256; i += 16) hist[row * 256 + i] = 0u;
    __syncthreads();
    unsigned pref = rowPref[row];
    int sh = pass * 8;
    for (int c2 = sub; c2 < S_; c2 += 16) {
      unsigned key = fkey(sc[row * SLD + c2]);
      if ((pass == 3) || ((key >> (sh + 8)) == pref))
        atomicAdd(&hist[row * 256 + ((key >> sh) & 0xFFu)], 1u);
    }
    __syncthreads();
    if (sub == 0) {
      unsigned krem = rowKrem[row], cum = 0;
      int bin;
      for (bin = 255; bin >= 0; --bin) {
        unsigned cnt = hist[row * 256 + bin];
        if (cum + cnt >= krem) break;
        cum += cnt;
      }
      if (bin < 0) bin = 0;
      rowKrem[row] = krem - cum;
      rowPref[row] = (pref << 8) | (unsigned)bin;
    }
    __syncthreads();
  }
  unsigned kth = rowPref[row];  // exact 819th-largest key (ties kept, s >= kth)

  float lm = -3.0e38f;
  for (int c2 = sub; c2 < S_; c2 += 16) lm = fmaxf(lm, sc[row * SLD + c2]);
  float* red = (float*)hist;    // reuse histogram space as reduction scratch
  red[row * 16 + sub] = lm;
  __syncthreads();
  if (sub == 0) {
    float m = red[row * 16];
    for (int i = 1; i < 16; ++i) m = fmaxf(m, red[row * 16 + i]);
    rowMax[row] = m;
  }
  __syncthreads();
  float mx = rowMax[row];
  float ls = 0.0f;
  for (int c2 = sub; c2 < S_; c2 += 16) {
    float s = sc[row * SLD + c2];
    float p = (fkey(s) >= kth) ? __expf(s - mx) : 0.0f;
    p16[row * PLD + c2] = (_Float16)p;      // packed f16 probs for WMMA A
    ls += p;
  }
  red[row * 16 + sub] = ls;
  __syncthreads();
  if (sub == 0) {
    float s = 0.0f;
    for (int i = 0; i < 16; ++i) s += red[row * 16 + i];
    rowSum[row] = s;
  }
  __syncthreads();

  // ---- phase 3: O = P x V, wave wv owns a 16-wide e-tile, 1/sum in epilogue -
  const _Float16* Vrow = vT16 + ((long)bh * HD_ + wv * 16 + (lane & 15)) * S_;
  f8 acc = {};
  int r2 = lane & 15, hf = lane >> 4;
  for (int kt = 0; kt < 64; ++kt) {
    h16 af;
    int k0 = kt * 32 + hf * 16;
#pragma unroll
    for (int j = 0; j < 16; ++j) {
      af[j]     = p16[r2 * PLD + k0 + j];
    }
    h16 bf;
    int t0 = kt * 32 + hf * 16;
#pragma unroll
    for (int j = 0; j < 16; ++j) bf[j] = Vrow[t0 + j];
    acc = wmma(af, bf, acc);
  }
  int n = lane & 15, m0 = (lane >> 4) * 8;
  long iidx = (long)h * HD_ + wv * 16 + n;    // channel index in [0,D)
#pragma unroll
  for (int r = 0; r < 8; ++r) {
    int m = m0 + r;
    float o = acc[r] / rowSum[m];
    out_t16[((long)b * D_ + iidx) * S_ + qbase + m] = (_Float16)o;  // [B,D,S]
  }
}

// ---------------------------------------------------------------------------
// Output projection: out[b,i,j] = sum_s out_t[b,i,s] * Wo[j,s] + bo[j]
// One wave per 16x64 tile, K=2048.
__global__ void __launch_bounds__(128) out_proj(const _Float16* __restrict__ ot16,
                                                const _Float16* __restrict__ wo16,
                                                const float* __restrict__ bo,
                                                float* __restrict__ out) {
  int lane = threadIdx.x & 31;
  int wid  = blockIdx.x * (blockDim.x >> 5) + (threadIdx.x >> 5);
  int b   = wid / ((D_ / 16) * (D_ / 64));
  int rem = wid % ((D_ / 16) * (D_ / 64));
  int it = rem / (D_ / 64);
  int jt = rem % (D_ / 64);

  const _Float16* Ap = ot16 + ((long)b * D_ + (long)it * 16) * S_;
  f8 acc[4];
#pragma unroll
  for (int s4 = 0; s4 < 4; ++s4)
#pragma unroll
    for (int j = 0; j < 8; ++j) acc[s4][j] = 0.0f;

  for (int kt = 0; kt < 64; ++kt) {
    h16 a = ldA(Ap, S_, lane, kt * 32);
#pragma unroll
    for (int s4 = 0; s4 < 4; ++s4) {
      const _Float16* Bp = wo16 + (long)(jt * 64 + s4 * 16) * D_;
      h16 bf = ldB(Bp, D_, lane, kt * 32);
      acc[s4] = wmma(a, bf, acc[s4]);
    }
  }
  int n = lane & 15, m0 = (lane >> 4) * 8;
#pragma unroll
  for (int s4 = 0; s4 < 4; ++s4) {
#pragma unroll
    for (int r = 0; r < 8; ++r) {
      int i = it * 16 + m0 + r;
      int j = jt * 64 + s4 * 16 + n;
      out[((long)b * D_ + i) * S_ + j] = acc[s4][r] + bo[j];
    }
  }
}

// ---------------------------------------------------------------------------
extern "C" void kernel_launch(void* const* d_in, const int* in_sizes, int n_in,
                              void* d_out, int out_size, void* d_ws, size_t ws_size,
                              hipStream_t stream) {
  (void)in_sizes; (void)n_in; (void)out_size; (void)ws_size;
  const float* x  = (const float*)d_in[0];
  // d_in[1] = causal_mask (bool) -- causality computed analytically, unused
  const float* Wq = (const float*)d_in[2];
  const float* Wk = (const float*)d_in[3];
  const float* Wv = (const float*)d_in[4];
  const float* bq = (const float*)d_in[5];
  const float* bk = (const float*)d_in[6];
  const float* bv = (const float*)d_in[7];
  const float* Wo = (const float*)d_in[8];
  const float* bo = (const float*)d_in[9];

  char* p = (char*)d_ws;
  auto carve = [&](size_t elems) -> _Float16* {
    _Float16* r = (_Float16*)p;
    p += ((elems * sizeof(_Float16)) + 255) & ~(size_t)255;
    return r;
  };
  const size_t WSZ = (size_t)H_ * HD_ * HD_;
  _Float16* x16  = carve((size_t)B_ * S_ * D_);        // 16 MB
  _Float16* w16  = carve(3 * WSZ);                     // 1.5 MB (Wq|Wk|Wv)
  _Float16* wo16 = carve((size_t)D_ * D_);             // 8 MB
  _Float16* q16  = carve((size_t)B_ * H_ * S_ * HD_);  // 16 MB
  _Float16* k16  = carve((size_t)B_ * H_ * S_ * HD_);  // 16 MB
  _Float16* vT16 = carve((size_t)B_ * H_ * S_ * HD_);  // 16 MB (transposed [e][s])
  _Float16* ot16 = carve((size_t)B_ * D_ * S_);        // 16 MB (transposed [i][s])

  cvt_f32_f16<<<4096, 256, 0, stream>>>(x, x16, B_ * S_ * D_);
  cvt_f32_f16<<<512, 256, 0, stream>>>(Wq, w16 + 0 * WSZ, (int)WSZ);
  cvt_f32_f16<<<512, 256, 0, stream>>>(Wk, w16 + 1 * WSZ, (int)WSZ);
  cvt_f32_f16<<<512, 256, 0, stream>>>(Wv, w16 + 2 * WSZ, (int)WSZ);
  cvt_f32_f16<<<4096, 256, 0, stream>>>(Wo, wo16, D_ * D_);

  // 4096 row tiles total, 4 waves/block
  qkv_proj<<<1024, 128, 0, stream>>>(x16, w16, bq, bk, bv, q16, k16, vT16);

  size_t smem = (size_t)16 * SLD * 4    // scores fp32
              + (size_t)16 * 256 * 4   // histograms
              + 64 * 4                 // row stats
              + (size_t)16 * PLD * 2;  // f16 probs
  sparse_attn<<<B_ * H_ * (S_ / 16), 256, smem, stream>>>(q16, k16, vT16, ot16);

  // 8192 tiles (B * 128 * 32), 4 waves/block
  out_proj<<<2048, 128, 0, stream>>>(ot16, wo16, bo, (float*)d_out);
}
